// GNNRegressor_70454643523899
// MI455X (gfx1250) — matrix operand order repaired and checked by reading, not compiled, not run on hardware
//
#include <hip/hip_runtime.h>

// ---------------- constants ----------------
#define NN 40000   // nodes
#define NE 640000  // edges
#define HH 128     // hidden
#define GG 128     // graphs

typedef __attribute__((ext_vector_type(16))) __bf16 v16bf;
typedef __attribute__((ext_vector_type(8)))  float  v8f;

__device__ __forceinline__ unsigned int f2bf1(float f) {
  unsigned int u = __float_as_uint(f);
  u += 0x7FFFu + ((u >> 16) & 1u);   // round-to-nearest-even
  return u >> 16;
}
__device__ __forceinline__ unsigned int packbf(float lo, float hi) {
  return f2bf1(lo) | (f2bf1(hi) << 16);
}

union FragBF { v16bf v; uint4 q[2]; unsigned int u[8]; };

// ---------------- weight prep: fp32 [K][N] -> bf16 [N][K] ----------------
__global__ void k_transpose_bf16(const float* __restrict__ src,
                                 unsigned short* __restrict__ dst,
                                 int K, int Nn) {
  int i = blockIdx.x * 256 + threadIdx.x;
  if (i >= K * Nn) return;
  int n = i / K, k = i - n * K;
  dst[i] = (unsigned short)f2bf1(src[k * Nn + n]);
}

__global__ void k_zero(float* __restrict__ p, int n) {
  int i = blockIdx.x * 256 + threadIdx.x;
  if (i < n) p[i] = 0.0f;
}

// ---------------- edge kernel ----------------
// Per wave: 16 edges. e = attr@We + be via WMMA (K padded 16->32),
// m = relu(x[src] + e), atomic scatter-add into agg[dst].
__global__ void __launch_bounds__(256) k_edge(
    const float* __restrict__ attr,               // E x 16
    const int* __restrict__ srcIdx, const int* __restrict__ dstIdx,
    const float* __restrict__ x, int ind,         // N x ind
    const unsigned short* __restrict__ WeT,       // [ind][16] bf16
    const float* __restrict__ be,                 // ind
    float* __restrict__ agg)                      // N x ind
{
  __shared__ alignas(16) unsigned short sW[128 * 16];
  __shared__ alignas(16) unsigned short sZ[16];   // 32B of zeros for B pad half
  __shared__ float sB[128];
  __shared__ int sS[128], sD[128];
  int tid = threadIdx.x;
  int e0 = blockIdx.x * 128;
  for (int i = tid; i < ind * 16; i += 256) sW[i] = WeT[i];
  for (int i = tid; i < ind; i += 256) sB[i] = be[i];
  if (tid < 16) sZ[tid] = 0;
  if (tid < 128) { sS[tid] = srcIdx[e0 + tid]; sD[tid] = dstIdx[e0 + tid]; }
  __syncthreads();

  int wave = tid >> 5, lane = tid & 31, hi = lane >> 4, lo = lane & 15;
  int we0 = wave * 16;
  unsigned uind = (unsigned)ind;

  // A: row M=lo (edge we0+lo), K = hi*8 + 0..7 real, K>=16 zero-pad
  FragBF A;
  {
    const float* ar = attr + (size_t)(e0 + we0 + lo) * 16 + hi * 8;
    float4 f0 = *(const float4*)(ar);
    float4 f1 = *(const float4*)(ar + 4);
    A.u[0] = packbf(f0.x, f0.y);
    A.u[1] = packbf(f0.z, f0.w);
    A.u[2] = packbf(f1.x, f1.y);
    A.u[3] = packbf(f1.z, f1.w);
    A.u[4] = A.u[5] = A.u[6] = A.u[7] = 0;
  }

  // hoisted per-row node offsets (nt-invariant), 32-bit unsigned for saddr form
  unsigned sOff[8], dOff[8];
#pragma unroll
  for (int r = 0; r < 8; ++r) {                  // C: VGPR r -> M = r + 8*hi
    int m = we0 + r + hi * 8;
    sOff[r] = (unsigned)sS[m] * uind;
    dOff[r] = (unsigned)sD[m] * uind;
  }

  int ntc = ind >> 4;
  for (int nt = 0; nt < ntc; ++nt) {
    unsigned col = (unsigned)(nt * 16 + lo);
    // B: elements 0-7 hold K=hi*8+0..7 (real); elements 8-15 (K>=16) -> zeros
    const uint4* wp = (const uint4*)&sW[col * 16 + hi * 8];
    FragBF B;
    B.q[0] = wp[0];
    B.q[1] = *(const uint4*)&sZ[0];
    v8f c = {};
    c = __builtin_amdgcn_wmma_f32_16x16x32_bf16(false, A.v, false, B.v,
                                                (short)0, c, false, false);
    float bias = sB[col];
    // batch the 8 gathers -> one wait, then 8 atomics
    float xv[8];
#pragma unroll
    for (int r = 0; r < 8; ++r) xv[r] = x[sOff[r] + col];
#pragma unroll
    for (int r = 0; r < 8; ++r) {
      float val = fmaxf(c[r] + bias + xv[r], 0.0f);
      unsafeAtomicAdd(&agg[dOff[r] + col], val);
    }
  }
}

// ---------------- node kernel: fused MLP + BN + relu ----------------
// h = x + agg; t = relu(h@W1+b1); o = t@W2+b2; x' = relu(BN(o)).  In-place safe.
__global__ void __launch_bounds__(256) k_node(
    const float* __restrict__ x, int ind,
    const float* __restrict__ agg,
    const unsigned short* __restrict__ W1t,       // [128][ind] bf16
    const float* __restrict__ b1,
    const unsigned short* __restrict__ W2t,       // [128][128] bf16
    const float* __restrict__ b2,
    const float* __restrict__ gmma, const float* __restrict__ beta,
    const float* __restrict__ mean, const float* __restrict__ var,
    float* __restrict__ xout)                     // N x 128
{
  const int TS = 136;                             // padded LDS row stride (shorts)
  __shared__ alignas(16) unsigned short tL[64 * 136];

  int tid = threadIdx.x, wave = tid >> 5, lane = tid & 31;
  int hi = lane >> 4, lo = lane & 15;
  int mtile = wave & 3, nclu = wave >> 2;
  int m0 = blockIdx.x * 64;
  unsigned uind = (unsigned)ind;

  int kcc = ind >> 5;                             // 2 (ind=64) or 4 (ind=128)
  FragBF Afr[4];
  {
    unsigned row = (unsigned)(m0 + mtile * 16 + lo);
    const float* xr = x + row * uind;
    const float* gr = agg + row * uind;
    for (int kc = 0; kc < kcc; ++kc) {
      int kb = kc * 32 + hi * 8;
      float4 x0 = *(const float4*)(xr + kb);      float4 g0 = *(const float4*)(gr + kb);
      float4 x1 = *(const float4*)(xr + kb + 4);  float4 g1 = *(const float4*)(gr + kb + 4);
      Afr[kc].u[0] = packbf(x0.x + g0.x, x0.y + g0.y);
      Afr[kc].u[1] = packbf(x0.z + g0.z, x0.w + g0.w);
      Afr[kc].u[2] = packbf(x1.x + g1.x, x1.y + g1.y);
      Afr[kc].u[3] = packbf(x1.z + g1.z, x1.w + g1.w);
      float4 x2 = *(const float4*)(xr + kb + 16); float4 g2 = *(const float4*)(gr + kb + 16);
      float4 x3 = *(const float4*)(xr + kb + 20); float4 g3 = *(const float4*)(gr + kb + 20);
      Afr[kc].u[4] = packbf(x2.x + g2.x, x2.y + g2.y);
      Afr[kc].u[5] = packbf(x2.z + g2.z, x2.w + g2.w);
      Afr[kc].u[6] = packbf(x3.x + g3.x, x3.y + g3.y);
      Afr[kc].u[7] = packbf(x3.z + g3.z, x3.w + g3.w);
    }
  }

  // phase 1: t = relu(h @ W1 + b1) -> LDS (bf16)
  for (int nti = 0; nti < 4; ++nti) {
    int nt = nclu * 4 + nti;
    unsigned col = (unsigned)(nt * 16 + lo);
    v8f c = {};
    for (int kc = 0; kc < kcc; ++kc) {
      FragBF B;
      const uint4* wp = (const uint4*)&W1t[col * uind + (unsigned)(kc * 32 + hi * 8)];
      B.q[0] = wp[0];
      B.q[1] = wp[2];
      c = __builtin_amdgcn_wmma_f32_16x16x32_bf16(false, Afr[kc].v, false, B.v,
                                                  (short)0, c, false, false);
    }
    float bias = b1[col];
#pragma unroll
    for (int r = 0; r < 8; ++r) {
      int row = mtile * 16 + r + hi * 8;
      float v = fmaxf(c[r] + bias, 0.0f);
      tL[row * TS + col] = (unsigned short)f2bf1(v);
    }
  }
  __syncthreads();

  // phase 2: out = t @ W2 + b2, BN, relu
  FragBF A2[4];
  {
    int row = mtile * 16 + lo;
#pragma unroll
    for (int kc = 0; kc < 4; ++kc) {
      int kb = kc * 32 + hi * 8;
      A2[kc].q[0] = *(const uint4*)&tL[row * TS + kb];
      A2[kc].q[1] = *(const uint4*)&tL[row * TS + kb + 16];
    }
  }
  for (int nti = 0; nti < 4; ++nti) {
    int nt = nclu * 4 + nti;
    unsigned col = (unsigned)(nt * 16 + lo);
    v8f c = {};
#pragma unroll
    for (int kc = 0; kc < 4; ++kc) {
      FragBF B;
      const uint4* wp = (const uint4*)&W2t[col * 128u + (unsigned)(kc * 32 + hi * 8)];
      B.q[0] = wp[0];
      B.q[1] = wp[2];
      c = __builtin_amdgcn_wmma_f32_16x16x32_bf16(false, A2[kc].v, false, B.v,
                                                  (short)0, c, false, false);
    }
    float bias = b2[col];
    float sc = gmma[col] * rsqrtf(var[col] + 1e-5f);
    float sh = beta[col] - mean[col] * sc;
#pragma unroll
    for (int r = 0; r < 8; ++r) {
      unsigned grow = (unsigned)(m0 + mtile * 16 + r + hi * 8);
      float v = (c[r] + bias) * sc + sh;
      xout[grow * 128u + col] = fmaxf(v, 0.0f);
    }
  }
}

// ---------------- pooling / heads ----------------
__global__ void k_pool(const float* __restrict__ x, const int* __restrict__ batch,
                       float* __restrict__ g) {
  unsigned i = blockIdx.x * 256 + threadIdx.x;
  if (i >= NN * 128u) return;
  unsigned node = i >> 7, col = i & 127u;
  unsafeAtomicAdd(&g[(unsigned)batch[node] * 128u + col], x[i]);
}

__global__ void k_head_z(const float* __restrict__ gsu,
                         const float* __restrict__ p1W, const float* __restrict__ p1b,
                         const float* __restrict__ p2W, const float* __restrict__ p2b,
                         float* __restrict__ zout) {
  __shared__ float sg[128], st[64];
  int g = blockIdx.x, j = threadIdx.x;
  sg[j] = gsu[g * 128 + j];
  __syncthreads();
  if (j < 64) {
    float acc = p1b[j];
    for (int k = 0; k < 128; ++k) acc += sg[k] * p1W[k * 64 + j];
    st[j] = fmaxf(acc, 0.0f);
  }
  __syncthreads();
  float acc = p2b[j];
  for (int k = 0; k < 64; ++k) acc += st[k] * p2W[k * 128 + j];
  zout[g * 128 + j] = 1.0f / (1.0f + expf(-acc));
}

__global__ void k_head_fc(const float* __restrict__ gsv, const float* __restrict__ gsu,
                          const float* __restrict__ gfeat,
                          const float* __restrict__ fcW, const float* __restrict__ fcb,
                          float* __restrict__ gfin, float* __restrict__ outgf) {
  __shared__ float sc[260];
  int g = blockIdx.x, j = threadIdx.x;
  sc[j] = gsv[g * 128 + j];
  sc[128 + j] = gsu[g * 128 + j];
  if (j < 4) sc[256 + j] = gfeat[g * 4 + j];
  __syncthreads();
  float acc = fcb[j];
  for (int k = 0; k < 260; ++k) acc += sc[k] * fcW[k * 128 + j];
  acc = fmaxf(acc, 0.0f);
  gfin[g * 128 + j] = acc;
  outgf[g * 128 + j] = acc;
}

__device__ __forceinline__ float softplusf(float x) {
  return x > 20.0f ? x : log1pf(expf(x));
}

__global__ void k_head_edl(const float* __restrict__ gfin,
                           const float* __restrict__ eW, const float* __restrict__ eb,
                           float* __restrict__ outedl) {
  int i = blockIdx.x * 256 + threadIdx.x;
  if (i >= GG * 4) return;
  int g = i >> 2, j = i & 3;
  float acc = eb[j];
  for (int k = 0; k < 128; ++k) acc += gfin[g * 128 + k] * eW[k * 4 + j];
  float v;
  if (j == 0)      v = acc;
  else if (j == 2) v = softplusf(acc) + 1.0f + 1e-6f;
  else             v = softplusf(acc) + 1e-6f;
  outedl[i] = v;
}

// ---------------- host launch ----------------
// Input indexing assumption: 10 top-level inputs in setup_inputs() dict order,
// then 'params' leaves in JAX pytree (alphabetical) order:
//   10 edl.W  11 edl.b  12 fc.W  13 fc.b  14 proj1.W 15 proj1.b 16 proj2.W 17 proj2.b
//   18..47 solute layers 0..2, 48..77 solvent layers 0..2; per layer:
//   +0 bn_beta +1 bn_gamma +2 bn_mean +3 bn_var +4 edge.W +5 edge.b
//   +6 lin1.W +7 lin1.b +8 lin2.W +9 lin2.b
extern "C" void kernel_launch(void* const* d_in, const int* in_sizes, int n_in,
                              void* d_out, int out_size, void* d_ws, size_t ws_size,
                              hipStream_t stream) {
  (void)in_sizes; (void)n_in; (void)out_size; (void)ws_size;
  const float* xin[2]  = { (const float*)d_in[0], (const float*)d_in[4] };
  const int*   eidx[2] = { (const int*)d_in[1],   (const int*)d_in[5] };
  const float* eat[2]  = { (const float*)d_in[2], (const float*)d_in[6] };
  const int*   bat[2]  = { (const int*)d_in[3],   (const int*)d_in[7] };
  const float* gfeat   = (const float*)d_in[8];
  const int pbase[2] = { 48, 18 };   // b=0: solvent, b=1: solute

  char* wsb = (char*)d_ws;
  size_t off = 0;
  auto alloc = [&](size_t bytes) -> char* {
    char* p = wsb + off;
    off = (off + bytes + 255) & ~(size_t)255;
    return p;
  };
  float* xbuf = (float*)alloc((size_t)NN * 128 * 4);
  float* agg  = (float*)alloc((size_t)NN * 128 * 4);
  float* gpool[2] = { (float*)alloc(GG * 128 * 4), (float*)alloc(GG * 128 * 4) };
  float* gfin = (float*)alloc(GG * 128 * 4);
  unsigned short *WeT[2][3], *W1t[2][3], *W2t[2][3];
  for (int b = 0; b < 2; ++b)
    for (int l = 0; l < 3; ++l) {
      int ind = l ? 128 : 64;
      WeT[b][l] = (unsigned short*)alloc((size_t)ind * 16 * 2);
      W1t[b][l] = (unsigned short*)alloc((size_t)128 * ind * 2);
      W2t[b][l] = (unsigned short*)alloc((size_t)128 * 128 * 2);
    }

  // weight prep (transpose + fp32->bf16)
  for (int b = 0; b < 2; ++b)
    for (int l = 0; l < 3; ++l) {
      int ind = l ? 128 : 64;
      int idx = pbase[b] + l * 10;
      const float* We = (const float*)d_in[idx + 4];
      const float* W1 = (const float*)d_in[idx + 6];
      const float* W2 = (const float*)d_in[idx + 8];
      int n1 = 16 * ind, n2 = ind * 128, n3 = 128 * 128;
      k_transpose_bf16<<<(n1 + 255) / 256, 256, 0, stream>>>(We, WeT[b][l], 16, ind);
      k_transpose_bf16<<<(n2 + 255) / 256, 256, 0, stream>>>(W1, W1t[b][l], ind, 128);
      k_transpose_bf16<<<(n3 + 255) / 256, 256, 0, stream>>>(W2, W2t[b][l], 128, 128);
    }

  // two backbones (reuse xbuf/agg: solvent pooled before solute starts)
  for (int b = 0; b < 2; ++b) {
    const float* xcur = xin[b];
    int ind = 64;
    for (int l = 0; l < 3; ++l) {
      int idx = pbase[b] + l * 10;
      int nz = NN * ind;
      k_zero<<<(nz + 255) / 256, 256, 0, stream>>>(agg, nz);
      k_edge<<<NE / 128, 256, 0, stream>>>(eat[b], eidx[b], eidx[b] + NE, xcur, ind,
                                           WeT[b][l], (const float*)d_in[idx + 5], agg);
      k_node<<<NN / 64, 256, 0, stream>>>(xcur, ind, agg,
                                          W1t[b][l], (const float*)d_in[idx + 7],
                                          W2t[b][l], (const float*)d_in[idx + 9],
                                          (const float*)d_in[idx + 1],   // gamma
                                          (const float*)d_in[idx + 0],   // beta
                                          (const float*)d_in[idx + 2],   // mean
                                          (const float*)d_in[idx + 3],   // var
                                          xbuf);
      xcur = xbuf; ind = 128;
    }
    int ng = GG * 128;
    k_zero<<<(ng + 255) / 256, 256, 0, stream>>>(gpool[b], ng);
    k_pool<<<(NN * 128) / 256, 256, 0, stream>>>(xbuf, bat[b], gpool[b]);
  }

  // heads; d_out = [edl (G*4) | z (G*128) | g_final (G*128)]
  float* out    = (float*)d_out;
  float* out_edl = out;
  float* out_z   = out + GG * 4;
  float* out_gf  = out + GG * 4 + GG * 128;
  k_head_z<<<GG, 128, 0, stream>>>(gpool[1],
      (const float*)d_in[14], (const float*)d_in[15],
      (const float*)d_in[16], (const float*)d_in[17], out_z);
  k_head_fc<<<GG, 128, 0, stream>>>(gpool[0], gpool[1], gfeat,
      (const float*)d_in[12], (const float*)d_in[13], gfin, out_gf);
  k_head_edl<<<2, 256, 0, stream>>>(gfin,
      (const float*)d_in[10], (const float*)d_in[11], out_edl);
}
